// InfoNCESep_17540646436979
// MI455X (gfx1250) — matrix operand order repaired and checked by reading, not compiled, not run on hardware
//
#include <hip/hip_runtime.h>
#include <hip/hip_bf16.h>

typedef __attribute__((ext_vector_type(2)))  float  v2f;
typedef __attribute__((ext_vector_type(8)))  float  v8f;
typedef __attribute__((ext_vector_type(8)))  __bf16 v8bf;
typedef __attribute__((ext_vector_type(16))) __bf16 v16bf;

#define N_ROWS    8192
#define DIM       512
#define ROWSTRIDE 1024   // V*D: stride between consecutive rows in the (N,2,512) inputs
#define INV_T     10.0f  // 1/TEMPERATURE
#define NSPLIT    8      // column-sweep split: 64 m-blocks x 8 n-chunks = 512 workgroups
#define COLS_PER_CHUNK (N_ROWS / NSPLIT)   // 1024

// ---------------- zero the scalar output (used by fallback's atomic path) ----------------
__global__ void zero_out_kernel(float* out) {
    if (threadIdx.x == 0 && blockIdx.x == 0) out[0] = 0.0f;
}

// ---------------- A: normalize + split f32 -> bf16 hi/lo ----------------
__global__ __launch_bounds__(256) void convA_kernel(const float* __restrict__ features,
                                                    __bf16* __restrict__ Ah,
                                                    __bf16* __restrict__ Al) {
    int row  = (blockIdx.x * blockDim.x + threadIdx.x) >> 5;
    int lane = threadIdx.x & 31;
    const float* src = features + (size_t)row * ROWSTRIDE;  // v=0 slice
    float x[16];
    float s = 0.0f;
    #pragma unroll
    for (int j = 0; j < 16; ++j) {
        x[j] = src[lane + 32 * j];
        s = fmaf(x[j], x[j], s);
    }
    #pragma unroll
    for (int m = 16; m >= 1; m >>= 1) s += __shfl_xor(s, m, 32);
    float inv = rsqrtf(s);
    #pragma unroll
    for (int j = 0; j < 16; ++j) {
        float  v  = x[j] * inv;
        __bf16 hi = (__bf16)v;
        __bf16 lo = (__bf16)(v - (float)hi);
        Ah[(size_t)row * DIM + lane + 32 * j] = hi;
        Al[(size_t)row * DIM + lane + 32 * j] = lo;
    }
}

// ---------------- B: split f32 -> bf16 hi/lo ----------------
__global__ __launch_bounds__(256) void convB_kernel(const float* __restrict__ f1,
                                                    __bf16* __restrict__ Bh,
                                                    __bf16* __restrict__ Bl) {
    size_t idx = (size_t)blockIdx.x * 256 + threadIdx.x;   // over 8192*512
    size_t i = idx >> 9;
    size_t k = idx & 511;
    float  v  = f1[i * ROWSTRIDE + k];                     // v=0 slice
    __bf16 hi = (__bf16)v;
    __bf16 lo = (__bf16)(v - (float)hi);
    Bh[idx] = hi;
    Bl[idx] = lo;
}

// ---------------- main: bf16x3 split GEMM fused with online logsumexp ----------------
// grid = dim3(64, NSPLIT): blockIdx.x = 128-row m-block, blockIdx.y = 1024-col n-chunk.
// 8 waves/WG; each wave owns a 16-row tile, sweeps its chunk's columns in groups of 64
// (4 accumulators). Per 32-wide K chunk per 16-col tile: 3 bf16 WMMAs (hi*hi+hi*lo+lo*hi).
// Per-row partial (m, s, pos) goes to workspace; merged by reduce kernels below.
__global__ __launch_bounds__(256) void infonce_bf16_kernel(const __bf16* __restrict__ Ah,
                                                           const __bf16* __restrict__ Al,
                                                           const __bf16* __restrict__ Bh,
                                                           const __bf16* __restrict__ Bl,
                                                           float* __restrict__ Pm,
                                                           float* __restrict__ Ps,
                                                           float* __restrict__ Pp) {
    const int lane = threadIdx.x & 31;
    const int waveId = threadIdx.x >> 5;
    const int r = lane & 15;   // A-fragment row / B-fragment column / C column
    const int h = lane >> 4;   // K-half (A/B), M-half (C/D)
    const int m0 = blockIdx.x * 128 + waveId * 16;
    const int chunk = blockIdx.y;
    const int c0 = chunk * COLS_PER_CHUNK;

    const __bf16* aRowH = Ah + (size_t)(m0 + r) * DIM;
    const __bf16* aRowL = Al + (size_t)(m0 + r) * DIM;

    float m_run[8], s_run[8], pos[8];
    #pragma unroll
    for (int v = 0; v < 8; ++v) { m_run[v] = -3.0e38f; s_run[v] = 0.0f; pos[v] = 0.0f; }

    for (int n0 = c0; n0 < c0 + COLS_PER_CHUNK; n0 += 64) {
        v8f acc0 = {}, acc1 = {}, acc2 = {}, acc3 = {};
        const __bf16* bRowH = Bh + (size_t)(n0 + r) * DIM + 16 * h;
        const __bf16* bRowL = Bl + (size_t)(n0 + r) * DIM + 16 * h;

        #pragma unroll 2
        for (int kc = 0; kc < DIM; kc += 32) {
            // A fragment: K = kc+16h+{0..15} contiguous
            v8bf ah0 = *(const v8bf*)(aRowH + kc + 16 * h);
            v8bf ah1 = *(const v8bf*)(aRowH + kc + 16 * h + 8);
            v8bf al0 = *(const v8bf*)(aRowL + kc + 16 * h);
            v8bf al1 = *(const v8bf*)(aRowL + kc + 16 * h + 8);
            v16bf ah = __builtin_shufflevector(ah0, ah1, 0,1,2,3,4,5,6,7,8,9,10,11,12,13,14,15);
            v16bf al = __builtin_shufflevector(al0, al1, 0,1,2,3,4,5,6,7,8,9,10,11,12,13,14,15);

            // B fragments: column n0+16t+r, K = kc+16h+{0..15} contiguous
            {
                v16bf bh = *(const v16bf*)(bRowH + (size_t)0 * 16 * DIM + kc);
                v16bf bl = *(const v16bf*)(bRowL + (size_t)0 * 16 * DIM + kc);
                acc0 = __builtin_amdgcn_wmma_f32_16x16x32_bf16(false, ah, false, bh, (short)0, acc0, false, false);
                acc0 = __builtin_amdgcn_wmma_f32_16x16x32_bf16(false, ah, false, bl, (short)0, acc0, false, false);
                acc0 = __builtin_amdgcn_wmma_f32_16x16x32_bf16(false, al, false, bh, (short)0, acc0, false, false);
            }
            {
                v16bf bh = *(const v16bf*)(bRowH + (size_t)1 * 16 * DIM + kc);
                v16bf bl = *(const v16bf*)(bRowL + (size_t)1 * 16 * DIM + kc);
                acc1 = __builtin_amdgcn_wmma_f32_16x16x32_bf16(false, ah, false, bh, (short)0, acc1, false, false);
                acc1 = __builtin_amdgcn_wmma_f32_16x16x32_bf16(false, ah, false, bl, (short)0, acc1, false, false);
                acc1 = __builtin_amdgcn_wmma_f32_16x16x32_bf16(false, al, false, bh, (short)0, acc1, false, false);
            }
            {
                v16bf bh = *(const v16bf*)(bRowH + (size_t)2 * 16 * DIM + kc);
                v16bf bl = *(const v16bf*)(bRowL + (size_t)2 * 16 * DIM + kc);
                acc2 = __builtin_amdgcn_wmma_f32_16x16x32_bf16(false, ah, false, bh, (short)0, acc2, false, false);
                acc2 = __builtin_amdgcn_wmma_f32_16x16x32_bf16(false, ah, false, bl, (short)0, acc2, false, false);
                acc2 = __builtin_amdgcn_wmma_f32_16x16x32_bf16(false, al, false, bh, (short)0, acc2, false, false);
            }
            {
                v16bf bh = *(const v16bf*)(bRowH + (size_t)3 * 16 * DIM + kc);
                v16bf bl = *(const v16bf*)(bRowL + (size_t)3 * 16 * DIM + kc);
                acc3 = __builtin_amdgcn_wmma_f32_16x16x32_bf16(false, ah, false, bh, (short)0, acc3, false, false);
                acc3 = __builtin_amdgcn_wmma_f32_16x16x32_bf16(false, ah, false, bl, (short)0, acc3, false, false);
                acc3 = __builtin_amdgcn_wmma_f32_16x16x32_bf16(false, al, false, bh, (short)0, acc3, false, false);
            }
        }

        #pragma unroll
        for (int v = 0; v < 8; ++v) {
            const int row = m0 + v + 8 * h;
            float x0 = acc0[v] * INV_T;
            float x1 = acc1[v] * INV_T;
            float x2 = acc2[v] * INV_T;
            float x3 = acc3[v] * INV_T;
            if (n0 + 0  + r == row) pos[v] = x0;
            if (n0 + 16 + r == row) pos[v] = x1;
            if (n0 + 32 + r == row) pos[v] = x2;
            if (n0 + 48 + r == row) pos[v] = x3;
            float gm    = fmaxf(fmaxf(x0, x1), fmaxf(x2, x3));
            float m_new = fmaxf(m_run[v], gm);
            float scale = __expf(m_run[v] - m_new);
            s_run[v] = s_run[v] * scale + __expf(x0 - m_new) + __expf(x1 - m_new)
                                        + __expf(x2 - m_new) + __expf(x3 - m_new);
            m_run[v] = m_new;
        }
    }

    // merge across the 16 lanes of each half, then lane r==v stores row partials
    #pragma unroll
    for (int v = 0; v < 8; ++v) {
        float m = m_run[v], s = s_run[v], p = pos[v];
        #pragma unroll
        for (int mask = 1; mask < 16; mask <<= 1) {
            float mo = __shfl_xor(m, mask, 32);
            float so = __shfl_xor(s, mask, 32);
            float po = __shfl_xor(p, mask, 32);
            float mn = fmaxf(m, mo);
            s = s * __expf(m - mn) + so * __expf(mo - mn);
            m = mn;
            p += po;
        }
        if (r == v) {
            const size_t slot = (size_t)chunk * N_ROWS + (m0 + v + 8 * h);
            Pm[slot] = m;
            Ps[slot] = s;
            Pp[slot] = p;
        }
    }
}

// ---------------- merge NSPLIT chunk partials per row -> per-row loss ----------------
__global__ __launch_bounds__(256) void reduce_kernel(const float* __restrict__ Pm,
                                                     const float* __restrict__ Ps,
                                                     const float* __restrict__ Pp,
                                                     float* __restrict__ rowLoss) {
    int row = blockIdx.x * 256 + threadIdx.x;   // 8192 rows
    float m = -3.0e38f, s = 0.0f, p = 0.0f;
    #pragma unroll
    for (int c = 0; c < NSPLIT; ++c) {
        size_t slot = (size_t)c * N_ROWS + row;
        float mc = Pm[slot], sc = Ps[slot], pc = Pp[slot];
        float mn = fmaxf(m, mc);
        s = s * __expf(m - mn) + sc * __expf(mc - mn);
        m = mn;
        p += pc;   // exactly one chunk holds the diagonal; others contribute 0
    }
    rowLoss[row] = (m + __logf(s)) - p;
}

// ---------------- deterministic single-block final mean ----------------
__global__ __launch_bounds__(256) void final_sum_kernel(const float* __restrict__ rowLoss,
                                                        float* __restrict__ out) {
    __shared__ float buf[256];
    float s = 0.0f;
    for (int i = threadIdx.x; i < N_ROWS; i += 256) s += rowLoss[i];
    buf[threadIdx.x] = s;
    __syncthreads();
    for (int st = 128; st > 0; st >>= 1) {
        if (threadIdx.x < st) buf[threadIdx.x] += buf[threadIdx.x + st];
        __syncthreads();
    }
    if (threadIdx.x == 0) out[0] = buf[0] * (1.0f / (float)N_ROWS);
}

// ================= fallback: exact-f32 WMMA path (needs only 32KB ws) =================
__global__ __launch_bounds__(256) void norm_kernel(const float* __restrict__ features,
                                                   float* __restrict__ inv_norm) {
    int row  = (blockIdx.x * blockDim.x + threadIdx.x) >> 5;
    int lane = threadIdx.x & 31;
    const float* r = features + (size_t)row * ROWSTRIDE;
    float s = 0.0f;
    #pragma unroll
    for (int k = 0; k < DIM / 32; ++k) { float x = r[lane + 32 * k]; s = fmaf(x, x, s); }
    #pragma unroll
    for (int m = 16; m >= 1; m >>= 1) s += __shfl_xor(s, m, 32);
    if (lane == 0) inv_norm[row] = rsqrtf(s);
}

__global__ __launch_bounds__(256) void infonce_f32_kernel(const float* __restrict__ anchor,
                                                          const float* __restrict__ f1,
                                                          const float* __restrict__ inv_norm,
                                                          float* __restrict__ out) {
    const int lane = threadIdx.x & 31;
    const int waveId = threadIdx.x >> 5;
    const int r = lane & 15;
    const int h = lane >> 4;
    const int m0 = blockIdx.x * 128 + waveId * 16;
    const float* aBase = anchor + (size_t)(m0 + r) * ROWSTRIDE + 2 * h;

    float m_run[8], s_run[8], pos[8], inorm[8];
    #pragma unroll
    for (int v = 0; v < 8; ++v) {
        m_run[v] = -3.0e38f; s_run[v] = 0.0f; pos[v] = 0.0f;
        inorm[v] = inv_norm[m0 + v + 8 * h] * INV_T;
    }

    for (int n0 = 0; n0 < N_ROWS; n0 += 64) {
        v8f acc0 = {}, acc1 = {}, acc2 = {}, acc3 = {};
        const float* bBase = f1 + (size_t)(n0 + r) * ROWSTRIDE + 2 * h;
        #pragma unroll 4
        for (int k = 0; k < DIM; k += 4) {
            v2f a  = *(const v2f*)(aBase + k);
            v2f b0 = *(const v2f*)(bBase + (size_t)0 * 16 * ROWSTRIDE + k);
            v2f b1 = *(const v2f*)(bBase + (size_t)1 * 16 * ROWSTRIDE + k);
            v2f b2 = *(const v2f*)(bBase + (size_t)2 * 16 * ROWSTRIDE + k);
            v2f b3 = *(const v2f*)(bBase + (size_t)3 * 16 * ROWSTRIDE + k);
            acc0 = __builtin_amdgcn_wmma_f32_16x16x4_f32(false, a, false, b0, (short)0, acc0, false, false);
            acc1 = __builtin_amdgcn_wmma_f32_16x16x4_f32(false, a, false, b1, (short)0, acc1, false, false);
            acc2 = __builtin_amdgcn_wmma_f32_16x16x4_f32(false, a, false, b2, (short)0, acc2, false, false);
            acc3 = __builtin_amdgcn_wmma_f32_16x16x4_f32(false, a, false, b3, (short)0, acc3, false, false);
        }
        #pragma unroll
        for (int v = 0; v < 8; ++v) {
            const int row = m0 + v + 8 * h;
            float x0 = acc0[v] * inorm[v];
            float x1 = acc1[v] * inorm[v];
            float x2 = acc2[v] * inorm[v];
            float x3 = acc3[v] * inorm[v];
            if (n0 + 0  + r == row) pos[v] = x0;
            if (n0 + 16 + r == row) pos[v] = x1;
            if (n0 + 32 + r == row) pos[v] = x2;
            if (n0 + 48 + r == row) pos[v] = x3;
            float gm    = fmaxf(fmaxf(x0, x1), fmaxf(x2, x3));
            float m_new = fmaxf(m_run[v], gm);
            float scale = __expf(m_run[v] - m_new);
            s_run[v] = s_run[v] * scale + __expf(x0 - m_new) + __expf(x1 - m_new)
                                        + __expf(x2 - m_new) + __expf(x3 - m_new);
            m_run[v] = m_new;
        }
    }

    float total = 0.0f;
    #pragma unroll
    for (int v = 0; v < 8; ++v) {
        float m = m_run[v], s = s_run[v], p = pos[v];
        #pragma unroll
        for (int mask = 1; mask < 16; mask <<= 1) {
            float mo = __shfl_xor(m, mask, 32);
            float so = __shfl_xor(s, mask, 32);
            float po = __shfl_xor(p, mask, 32);
            float mn = fmaxf(m, mo);
            s = s * __expf(m - mn) + so * __expf(mo - mn);
            m = mn;
            p += po;
        }
        total += (m + __logf(s)) - p;
    }
    if (r == 0) atomicAdd(out, total * (1.0f / (float)N_ROWS));
}

extern "C" void kernel_launch(void* const* d_in, const int* in_sizes, int n_in,
                              void* d_out, int out_size, void* d_ws, size_t ws_size,
                              hipStream_t stream) {
    const float* features  = (const float*)d_in[0];  // (8192, 2, 512) f32
    const float* features1 = (const float*)d_in[1];  // (8192, 2, 512) f32
    float* out = (float*)d_out;

    const size_t planeElems   = (size_t)N_ROWS * DIM;             // 4M bf16 elements/plane
    const size_t partialElems = (size_t)NSPLIT * N_ROWS;          // 64K floats/partial
    const size_t needBytes = 4 * planeElems * sizeof(__bf16)      // Ah/Al/Bh/Bl: 32 MB
                           + 3 * partialElems * sizeof(float)     // Pm/Ps/Pp:   768 KB
                           + (size_t)N_ROWS * sizeof(float);      // rowLoss:     32 KB

    if (ws_size >= needBytes) {
        __bf16* Ah = (__bf16*)d_ws;
        __bf16* Al = Ah + planeElems;
        __bf16* Bh = Al + planeElems;
        __bf16* Bl = Bh + planeElems;
        float*  Pm = (float*)(Bl + planeElems);
        float*  Ps = Pm + partialElems;
        float*  Pp = Ps + partialElems;
        float*  rowLoss = Pp + partialElems;

        convA_kernel<<<N_ROWS / 8, 256, 0, stream>>>(features, Ah, Al);
        convB_kernel<<<(int)(planeElems / 256), 256, 0, stream>>>(features1, Bh, Bl);
        infonce_bf16_kernel<<<dim3(N_ROWS / 128, NSPLIT), 256, 0, stream>>>(Ah, Al, Bh, Bl, Pm, Ps, Pp);
        reduce_kernel<<<N_ROWS / 256, 256, 0, stream>>>(Pm, Ps, Pp, rowLoss);
        final_sum_kernel<<<1, 256, 0, stream>>>(rowLoss, out);
    } else {
        float* inv_norm = (float*)d_ws;  // 32 KB
        zero_out_kernel<<<1, 32, 0, stream>>>(out);
        norm_kernel<<<N_ROWS / 8, 256, 0, stream>>>(features, inv_norm);
        infonce_f32_kernel<<<N_ROWS / 128, 256, 0, stream>>>(features, features1, inv_norm, out);
    }
}
